// GraphAttentionLayer_78185584656634
// MI455X (gfx1250) — matrix operand order repaired and checked by reading, not compiled, not run on hardware
//
#include <hip/hip_runtime.h>
#include <hip/hip_bf16.h>

typedef __bf16 bf16_t;
typedef __attribute__((ext_vector_type(16))) __bf16 v16bf;
typedef __attribute__((ext_vector_type(8)))  float  v8f;

#define N_NODES 8192
#define IN_FEAT 256
#define OUT_FEAT 128
#define WAVES_PER_BLOCK 4   // 128 threads, 4 query tiles of 16 rows per block

union BFrag {                 // 16 bf16 = 8 VGPRs
    v16bf v;
    uint4 u[2];
};

// ---------------------------------------------------------------------------
// Kernel 1: h = X @ W^T + b   (f32 compute, emit bf16 row-major AND transposed)
// ---------------------------------------------------------------------------
__global__ __launch_bounds__(OUT_FEAT)
void gat_linear_kernel(const float* __restrict__ X, const float* __restrict__ W,
                       const float* __restrict__ b,
                       bf16_t* __restrict__ hb, bf16_t* __restrict__ htb)
{
    __shared__ float xs[IN_FEAT];
    const int n = blockIdx.x;
    for (int d = threadIdx.x; d < IN_FEAT; d += blockDim.x)
        xs[d] = X[n * IN_FEAT + d];
    __syncthreads();

    const int o = threadIdx.x;
    const float* wr = W + o * IN_FEAT;
    float acc = b[o];
#pragma unroll 8
    for (int d = 0; d < IN_FEAT; ++d)
        acc += xs[d] * wr[d];

    const bf16_t v = (bf16_t)acc;
    hb [n * OUT_FEAT + o] = v;    // h  [node][feat]
    htb[o * N_NODES  + n] = v;    // h^T[feat][node]
}

// ---------------------------------------------------------------------------
// Kernel 2: t = h @ attn_weights   (f32 compute, emit bf16)
// ---------------------------------------------------------------------------
__global__ __launch_bounds__(OUT_FEAT)
void gat_tmat_kernel(const bf16_t* __restrict__ hb, const float* __restrict__ A,
                     bf16_t* __restrict__ tb)
{
    __shared__ float hs[OUT_FEAT];
    const int n = blockIdx.x;
    hs[threadIdx.x] = (float)hb[n * OUT_FEAT + threadIdx.x];
    __syncthreads();

    const int e = threadIdx.x;
    float acc = 0.0f;
#pragma unroll 8
    for (int d = 0; d < OUT_FEAT; ++d)
        acc += hs[d] * A[d * OUT_FEAT + e];

    tb[n * OUT_FEAT + e] = (bf16_t)acc;
}

// ---------------------------------------------------------------------------
// Async stage of one 32-key chunk into LDS (per block, all 128 threads).
// ---------------------------------------------------------------------------
__device__ __forceinline__
void issue_chunk_async(const bf16_t* __restrict__ hb, const bf16_t* __restrict__ htb,
                       bf16_t* hbuf, bf16_t* htbuf, int j0, int tid)
{
#pragma unroll
    for (int it = 0; it < 4; ++it) {                 // 512 x 16B tasks, h rows
        const int task = tid + 128 * it;
        const int row  = task >> 4;
        const int part = task & 15;
        const unsigned long long g =
            (unsigned long long)(uintptr_t)(hb + (size_t)(j0 + row) * OUT_FEAT) +
            (unsigned long long)(part * 16);
        const unsigned int l = (unsigned int)(uintptr_t)hbuf + (unsigned int)(task * 16);
        asm volatile("global_load_async_to_lds_b128 %0, %1, off"
                     :: "v"(l), "v"(g) : "memory");
    }
#pragma unroll
    for (int it = 0; it < 4; ++it) {                 // 512 x 16B tasks, h^T slice
        const int task = tid + 128 * it;
        const int f    = task >> 2;
        const int part = task & 3;
        const unsigned long long g =
            (unsigned long long)(uintptr_t)(htb + (size_t)f * N_NODES + j0) +
            (unsigned long long)(part * 16);
        const unsigned int l = (unsigned int)(uintptr_t)htbuf + (unsigned int)(task * 16);
        asm volatile("global_load_async_to_lds_b128 %0, %1, off"
                     :: "v"(l), "v"(g) : "memory");
    }
}

// 16-load LDS burst with a single trailing wait: 16 distinct uint4 destinations,
// one s_wait_dscnt for the whole group (defs complete before any consumer runs).
#define DS_BURST16_WAIT(a, o0,o1,o2,o3,o4,o5,o6,o7,o8,o9,o10,o11,o12,o13,o14,o15,     \
                        q0,q1,q2,q3,q4,q5,q6,q7,q8,q9,q10,q11,q12,q13,q14,q15)        \
    asm volatile("ds_load_b128 %0, %16 offset:" #o0 "\n\t"                            \
                 "ds_load_b128 %1, %16 offset:" #o1 "\n\t"                            \
                 "ds_load_b128 %2, %16 offset:" #o2 "\n\t"                            \
                 "ds_load_b128 %3, %16 offset:" #o3 "\n\t"                            \
                 "ds_load_b128 %4, %16 offset:" #o4 "\n\t"                            \
                 "ds_load_b128 %5, %16 offset:" #o5 "\n\t"                            \
                 "ds_load_b128 %6, %16 offset:" #o6 "\n\t"                            \
                 "ds_load_b128 %7, %16 offset:" #o7 "\n\t"                            \
                 "ds_load_b128 %8, %16 offset:" #o8 "\n\t"                            \
                 "ds_load_b128 %9, %16 offset:" #o9 "\n\t"                            \
                 "ds_load_b128 %10, %16 offset:" #o10 "\n\t"                          \
                 "ds_load_b128 %11, %16 offset:" #o11 "\n\t"                          \
                 "ds_load_b128 %12, %16 offset:" #o12 "\n\t"                          \
                 "ds_load_b128 %13, %16 offset:" #o13 "\n\t"                          \
                 "ds_load_b128 %14, %16 offset:" #o14 "\n\t"                          \
                 "ds_load_b128 %15, %16 offset:" #o15 "\n\t"                          \
                 "s_wait_dscnt 0x0"                                                   \
                 : "=v"(q0), "=v"(q1), "=v"(q2), "=v"(q3),                            \
                   "=v"(q4), "=v"(q5), "=v"(q6), "=v"(q7),                            \
                   "=v"(q8), "=v"(q9), "=v"(q10), "=v"(q11),                          \
                   "=v"(q12), "=v"(q13), "=v"(q14), "=v"(q15)                         \
                 : "v"(a) : "memory")

// ---------------------------------------------------------------------------
// Kernel 3: fused flash attention, LDS-staged key stream
// ---------------------------------------------------------------------------
__global__ __launch_bounds__(32 * WAVES_PER_BLOCK, 1)
void gat_flash_kernel(const bf16_t* __restrict__ hb,   // [N][128]
                      const bf16_t* __restrict__ htb,  // [128][N]
                      const bf16_t* __restrict__ tb,   // [N][128]
                      float* __restrict__ out)         // [N][128]
{
    __shared__ __align__(16) bf16_t hbuf [2][32 * OUT_FEAT];   // 2 x 8 KB
    __shared__ __align__(16) bf16_t htbuf[2][OUT_FEAT * 32];   // 2 x 8 KB

    const int tid  = threadIdx.x;
    const int lane = tid & 31;
    const int wave = tid >> 5;
    const int lm   = lane & 15;
    const int half = lane >> 4;
    const int i0   = (blockIdx.x * WAVES_PER_BLOCK + wave) * 16;  // query tile base

    // B operand for score GEMM: T^T fragments (resident all loop long)
    BFrag bq[4];
    {
        const bf16_t* trow = tb + (size_t)(i0 + lm) * OUT_FEAT;
#pragma unroll
        for (int kf = 0; kf < 4; ++kf) {
            const uint4* p = reinterpret_cast<const uint4*>(trow + kf * 32 + half * 8);
            bq[kf].u[0] = p[0];
            bq[kf].u[1] = p[2];
        }
    }

    const v8f vzero = {0.f, 0.f, 0.f, 0.f, 0.f, 0.f, 0.f, 0.f};
    v8f acc[8];
#pragma unroll
    for (int nt = 0; nt < 8; ++nt) acc[nt] = vzero;

    float row_m = -3.0e38f;
    float row_l = 0.0f;

    // prologue: stage chunk 0
    issue_chunk_async(hb, htb, hbuf[0], htbuf[0], 0, tid);

#pragma unroll 1
    for (int c = 0; c < N_NODES / 32; ++c) {
        const int buf = c & 1;

        asm volatile("s_wait_asynccnt 0x0" ::: "memory");  // my DMA done
        __syncthreads();          // everyone's DMA done AND everyone left buf^1

        if (c + 1 < N_NODES / 32)                          // overlap next chunk DMA
            issue_chunk_async(hb, htb, hbuf[buf ^ 1], htbuf[buf ^ 1], (c + 1) * 32, tid);

        // per-lane LDS bases for this chunk's fragments
        const unsigned aS = (unsigned)(uintptr_t)(&hbuf[buf][0]) +
                            (unsigned)(lm * 256 + half * 16);
        const unsigned aB = (unsigned)(uintptr_t)(&htbuf[buf][0]) +
                            (unsigned)(lm * 64 + half * 16);

        // ---- burst-load ALL 16 score A fragments, single wait ----
        uint4 q0,q1,q2,q3,q4,q5,q6,q7,q8,q9,q10,q11,q12,q13,q14,q15;
        DS_BURST16_WAIT(aS, 0,32,64,96,128,160,192,224,
                        4096,4128,4160,4192,4224,4256,4288,4320,
                        q0,q1,q2,q3,q4,q5,q6,q7,q8,q9,q10,q11,q12,q13,q14,q15);

        BFrag ah0[4], ah1[4];
        ah0[0].u[0]=q0;  ah0[0].u[1]=q1;  ah0[1].u[0]=q2;  ah0[1].u[1]=q3;
        ah0[2].u[0]=q4;  ah0[2].u[1]=q5;  ah0[3].u[0]=q6;  ah0[3].u[1]=q7;
        ah1[0].u[0]=q8;  ah1[0].u[1]=q9;  ah1[1].u[0]=q10; ah1[1].u[1]=q11;
        ah1[2].u[0]=q12; ah1[2].u[1]=q13; ah1[3].u[0]=q14; ah1[3].u[1]=q15;

        // ---- S^T tiles: two independent accumulator chains, no load stalls ----
        v8f c0 = vzero, c1 = vzero;
#pragma unroll
        for (int kf = 0; kf < 4; ++kf) {
            c0 = __builtin_amdgcn_wmma_f32_16x16x32_bf16(
                     false, ah0[kf].v, false, bq[kf].v, (short)0, c0, false, false);
            c1 = __builtin_amdgcn_wmma_f32_16x16x32_bf16(
                     false, ah1[kf].v, false, bq[kf].v, (short)0, c1, false, false);
        }
        float st[2][8];
#pragma unroll
        for (int r = 0; r < 8; ++r) { st[0][r] = c0[r]; st[1][r] = c1[r]; }

        // ---- burst-load aggregation B fragments, single wait ----
        uint4 p0,p1,p2,p3,p4,p5,p6,p7,p8,p9,p10,p11,p12,p13,p14,p15;
        DS_BURST16_WAIT(aB, 0,32,1024,1056,2048,2080,3072,3104,
                        4096,4128,5120,5152,6144,6176,7168,7200,
                        p0,p1,p2,p3,p4,p5,p6,p7,p8,p9,p10,p11,p12,p13,p14,p15);

        BFrag bhf[8];
        bhf[0].u[0]=p0;  bhf[0].u[1]=p1;  bhf[1].u[0]=p2;  bhf[1].u[1]=p3;
        bhf[2].u[0]=p4;  bhf[2].u[1]=p5;  bhf[3].u[0]=p6;  bhf[3].u[1]=p7;
        bhf[4].u[0]=p8;  bhf[4].u[1]=p9;  bhf[5].u[0]=p10; bhf[5].u[1]=p11;
        bhf[6].u[0]=p12; bhf[6].u[1]=p13; bhf[7].u[0]=p14; bhf[7].u[1]=p15;

        // ---- online softmax (per lane == query row lm; partner lane +16) ----
        float vmax = st[0][0];
#pragma unroll
        for (int jt = 0; jt < 2; ++jt)
#pragma unroll
            for (int r = 0; r < 8; ++r) vmax = fmaxf(vmax, st[jt][r]);
        vmax = fmaxf(vmax, __shfl_xor(vmax, 16, 32));

        const float mnew  = fmaxf(row_m, vmax);
        const float alpha = __expf(row_m - mnew);
        row_m = mnew;

        BFrag pf;
        float s = 0.0f;
#pragma unroll
        for (int jt = 0; jt < 2; ++jt)
#pragma unroll
            for (int r = 0; r < 8; ++r) {
                const float pe = __expf(st[jt][r] - mnew);
                s += pe;
                pf.v[jt * 8 + r] = (bf16_t)pe;
            }
        s += __shfl_xor(s, 16, 32);
        row_l = row_l * alpha + s;

        // rescale accumulators: acc row (r + 8*half) needs alpha of query (8*half+r)
#pragma unroll
        for (int r = 0; r < 8; ++r) {
            const float aq = __shfl(alpha, 8 * half + r, 32);
#pragma unroll
            for (int nt = 0; nt < 8; ++nt) acc[nt][r] *= aq;
        }

        // ---- O += P(16x32) x H(32x16) per output 16-feat tile ----
#pragma unroll
        for (int nt = 0; nt < 8; ++nt)
            acc[nt] = __builtin_amdgcn_wmma_f32_16x16x32_bf16(
                          false, pf.v, false, bhf[nt].v, (short)0, acc[nt], false, false);
    }

    // ---- finalize: divide by row sum, write f32 output ----
#pragma unroll
    for (int r = 0; r < 8; ++r) {
        const float lq  = __shfl(row_l, 8 * half + r, 32);
        const float inv = 1.0f / lq;
#pragma unroll
        for (int nt = 0; nt < 8; ++nt)
            out[(size_t)(i0 + 8 * half + r) * OUT_FEAT + nt * 16 + lm] = acc[nt][r] * inv;
    }
}

// ---------------------------------------------------------------------------
extern "C" void kernel_launch(void* const* d_in, const int* in_sizes, int n_in,
                              void* d_out, int out_size, void* d_ws, size_t ws_size,
                              hipStream_t stream)
{
    // inputs: [0]=adjacency (UNUSED by reference), [1]=node_features,
    //         [2]=W, [3]=b, [4]=attn_weights
    const float* X = (const float*)d_in[1];
    const float* W = (const float*)d_in[2];
    const float* b = (const float*)d_in[3];
    const float* A = (const float*)d_in[4];
    float* out = (float*)d_out;

    char* ws = (char*)d_ws;
    const size_t hbytes = (size_t)N_NODES * OUT_FEAT * sizeof(bf16_t);  // 2 MB
    bf16_t* hb  = (bf16_t*)(ws);               // h   [N][128] bf16
    bf16_t* htb = (bf16_t*)(ws + hbytes);      // h^T [128][N] bf16
    bf16_t* tb  = (bf16_t*)(ws + 2 * hbytes);  // t   [N][128] bf16

    gat_linear_kernel<<<N_NODES, OUT_FEAT, 0, stream>>>(X, W, b, hb, htb);
    gat_tmat_kernel<<<N_NODES, OUT_FEAT, 0, stream>>>(hb, A, tb);
    gat_flash_kernel<<<N_NODES / (16 * WAVES_PER_BLOCK), 32 * WAVES_PER_BLOCK, 0, stream>>>(
        hb, htb, tb, out);
}
